// SparseConv3d_1563368095916
// MI455X (gfx1250) — matrix operand order repaired and checked by reading, not compiled, not run on hardware
//
#include <hip/hip_runtime.h>

// ---------------------------------------------------------------------------
// SparseConv3d implicit GEMM for gfx1250 (MI455X), wave32 + WMMA f16->f32.
//   features : [N, 64]  fp16
//   indices  : [M, 27]  int  (-1 => missing neighbor)
//   weight   : [27, 64, 64] fp16   (k, cin, cout)
//   out      : [M, 64]  fp16  (fp32 accumulation inside)
//
// Gather of missing neighbors goes through a 128B zero page in scratch, so
// there is no per-offset divergent branch and no fragment zero-fill: just an
// address select (v_cndmask) + unconditional b128 loads.
// ---------------------------------------------------------------------------

typedef __attribute__((ext_vector_type(16))) _Float16 v16h;
typedef __attribute__((ext_vector_type(8)))  _Float16 v8h;
typedef __attribute__((ext_vector_type(8)))  float    v8f;

#define K3   27
#define CIN  64
#define COUT 64
#define WT_BYTES (K3 * CIN * COUT * 2)   // 221,184 B; zero page follows it

union Frag16 {
    v16h v;
    v8h  h[2];
};

// Pre-pass: transpose weights to wT[k][cout][cin] (makes B fragments two
// contiguous 16B loads per lane) and zero-fill the 128B zero page.
__global__ __launch_bounds__(256) void spconv_wtrans_kernel(
    const _Float16* __restrict__ w, _Float16* __restrict__ wT,
    _Float16* __restrict__ zpage) {
    int t = blockIdx.x * 256 + threadIdx.x;
    if (t < K3 * CIN * COUT) {
        int base = t & ~(CIN * COUT - 1);   // k * 4096
        int co   = (t >> 6) & 63;           // dest is [k][co][ci]
        int ci   = t & 63;
        wT[t] = w[base + ci * COUT + co];
    }
    if (t < CIN)
        zpage[t] = (_Float16)0;
}

// Main kernel: 256 threads = 8 waves; block computes a 32x64 output tile,
// each wave one 16x16 sub-tile, accumulating over 27 offsets x (CIN=64 as
// two K=32 WMMA steps) in fp32.
__global__ __launch_bounds__(256) void spconv_wmma_kernel(
    const _Float16* __restrict__ features,
    const int*      __restrict__ indices,
    const _Float16* __restrict__ wT,
    const _Float16* __restrict__ zpage,
    _Float16*       __restrict__ out,
    int M) {
    const int lane    = threadIdx.x & 31;
    const int wave    = threadIdx.x >> 5;
    const int halfsel = lane >> 4;     // 0: lanes 0-15, 1: lanes 16-31
    const int l16     = lane & 15;

    const int m0 = blockIdx.x * 32 + (wave >> 2) * 16;  // row sub-tile base
    const int n0 = (wave & 3) * 16;                     // col sub-tile base
    if (m0 >= M) return;  // uniform per wave (M multiple of 32 anyway)

    // ISA 16-bit A layout: lane holds row m=l16; halves 0..7 -> K=b..b+7,
    // halves 8..15 -> K=16+b..16+b+7, b = halfsel*8 (per 32-wide K chunk).
    const int aoff = halfsel * 8;
    // ISA 16-bit B layout: lane holds col n=l16; same K as A, b..b+7 and 16+b..16+b+7
    // (per 32-wide K chunk); B loads use aoff, koff is unused.
    const int koff = halfsel * 16;

    const int myrow = m0 + l16;
    const int mycol = n0 + l16;

    const _Float16* wb = wT + (size_t)mycol * CIN;  // + k*4096 + kc*32 + aoff (+16)
    const int*      ib = indices + (size_t)myrow * K3;

    v8f acc = {};

#pragma unroll
    for (int k = 0; k < K3; ++k) {
        const _Float16* wk = wb + k * (CIN * COUT);
        // prefetch next offset's weight row (global_prefetch_b8)
        if (k + 1 < K3)
            __builtin_prefetch(wk + CIN * COUT, 0, 1);

        // ---- gather A fragments: address select, no divergent branch ------
        int idx = ib[k];
        const _Float16* f =
            (idx >= 0) ? (features + (size_t)idx * CIN) : zpage;

        Frag16 a0, a1;
        a0.h[0] = *(const v8h*)(f + 0  + aoff);   // K =  b .. b+7
        a0.h[1] = *(const v8h*)(f + 16 + aoff);   // K = 16+b..16+b+7
        a1.h[0] = *(const v8h*)(f + 32 + aoff);   // chunk 1
        a1.h[1] = *(const v8h*)(f + 48 + aoff);

        // ---- load B fragments from transposed weights ---------------------
        Frag16 b0, b1;
        b0.h[0] = *(const v8h*)(wk + aoff);           // K = aoff..aoff+7
        b0.h[1] = *(const v8h*)(wk + 16 + aoff);      // K = 16+aoff..16+aoff+7
        b1.h[0] = *(const v8h*)(wk + 32 + aoff);      // chunk 1
        b1.h[1] = *(const v8h*)(wk + 48 + aoff);

        // ---- matrix cores: D = A*B + C, fp32 accumulate -------------------
        acc = __builtin_amdgcn_wmma_f32_16x16x32_f16(
            false, a0.v, false, b0.v, (short)0, acc, false, false);
        acc = __builtin_amdgcn_wmma_f32_16x16x32_f16(
            false, a1.v, false, b1.v, (short)0, acc, false, false);
    }

    // ---- epilogue: C/D layout -> out[m][n], cast fp32 -> fp16 -------------
    // VGPR v of C/D holds row m0 + v + halfsel*8, col = mycol.
    _Float16* o = out + (size_t)(m0 + halfsel * 8) * COUT + mycol;
#pragma unroll
    for (int v = 0; v < 8; ++v)
        o[(size_t)v * COUT] = (_Float16)acc[v];
}

extern "C" void kernel_launch(void* const* d_in, const int* in_sizes, int n_in,
                              void* d_out, int out_size, void* d_ws, size_t ws_size,
                              hipStream_t stream) {
    const _Float16* features = (const _Float16*)d_in[0];
    const int*      indices  = (const int*)d_in[1];
    const _Float16* weight   = (const _Float16*)d_in[2];
    _Float16*       out      = (_Float16*)d_out;
    _Float16*       wT       = (_Float16*)d_ws;
    _Float16*       zpage    = (_Float16*)((char*)d_ws + WT_BYTES); // 128 B

    const int M = in_sizes[1] / K3;

    // 1) transpose weights + zero page into scratch
    {
        int total  = K3 * CIN * COUT;
        int blocks = (total + 255) / 256;
        spconv_wtrans_kernel<<<blocks, 256, 0, stream>>>(weight, wT, zpage);
    }
    // 2) main WMMA kernel: one block per 32 output rows
    {
        int blocks = (M + 31) / 32;
        spconv_wmma_kernel<<<blocks, 256, 0, stream>>>(features, indices, wT,
                                                       zpage, out, M);
    }
}
